// Decoder1_32873679683850
// MI455X (gfx1250) — matrix operand order repaired and checked by reading, not compiled, not run on hardware
//
#include <hip/hip_runtime.h>
#include <hip/hip_bf16.h>
#include <stdint.h>

typedef __attribute__((ext_vector_type(16))) __bf16 v16bf;
typedef __attribute__((ext_vector_type(8)))  float  v8f;

#define T_STEPS 4
#define C_DIM   256
#define N_DIM   1024
#define TB_DIM  128
#define M_TOTAL (TB_DIM * N_DIM)        // 131072 GEMM rows
#define INV_M   (1.0f / (float)M_TOTAL)
#define BN_EPS  1e-5f

__device__ __forceinline__ unsigned pack_bf16(float a, float b) {
  union { float f; unsigned u; } ua, ub;
  ua.f = a; ub.f = b;
  unsigned x = ua.u, y = ub.u;
  x = (x + 0x7FFFu + ((x >> 16) & 1u)) >> 16;   // round-to-nearest-even
  y = (y + 0x7FFFu + ((y >> 16) & 1u)) >> 16;
  return (x & 0xFFFFu) | (y << 16);
}

__device__ __forceinline__ float bf16_to_f32(unsigned h) {
  union { unsigned u; float f; } c; c.u = h << 16; return c.f;
}

union FragBF { v16bf v; unsigned u[8]; uint4 q[2]; };

// ---------------------------------------------------------------------------
// Kernel 0: zero the BN-stat accumulators (sum[256], sumsq[256]) in d_ws.
// ---------------------------------------------------------------------------
__global__ void zero_stats(float* __restrict__ stats) {
  stats[blockIdx.x * 256 + threadIdx.x] = 0.0f;
}

// ---------------------------------------------------------------------------
// Kernel 1: GEMM Y = X * W^T (bf16 WMMA, f32 accumulate) + per-channel
// sum / sum-of-squares for BatchNorm, Y written bf16 into L2-resident ws.
// Grid: 1024 blocks x 256 threads (8 waves). Wave w owns M-tile
// (blockIdx*8 + w) of 16 rows x all 256 output channels.
// ---------------------------------------------------------------------------
__global__ __launch_bounds__(256, 1) void snn_gemm_bn_stats(
    const float* __restrict__ X, const float* __restrict__ W,
    float* __restrict__ stats, unsigned* __restrict__ Yws)
{
  // W staged as bf16 in exact B-fragment order: [otile][kchunk][lane][dword]
  // B 32x16 layout (ISA 7.12.2/7.12.4): lane<16 -> N=lane, K=0..7,16..23 in V0..7
  // (two 16b halves per VGPR); lane>=16 -> N=lane-16, K=8..15,24..31.
  __shared__ unsigned wlds[16][8][32][8];       // 128 KB
  __shared__ float s1[C_DIM];
  __shared__ float s2[C_DIM];

  const int tid = threadIdx.x;
  s1[tid] = 0.0f;
  s2[tid] = 0.0f;

  // Stage + swizzle W (256x256 fp32 -> bf16 pairs). 32768 dword slots.
  for (int p = tid; p < 32768; p += 256) {
    const int o = p >> 7;                       // W row (output channel)
    const int k = (p & 127) << 1;               // even K
    const float2 wv = *(const float2*)(W + o * C_DIM + k);
    const int kin  = k & 31;
    const int lane = (o & 15) | (((kin >> 3) & 1) << 4);
    const int dw   = ((kin >> 4) << 2) | ((kin & 7) >> 1);
    wlds[o >> 4][k >> 5][lane][dw] = pack_bf16(wv.x, wv.y);
  }
  __syncthreads();

  const int wave  = tid >> 5;
  const int lane  = tid & 31;
  const int mtile = blockIdx.x * 8 + wave;              // 0..8191
  const int mrow  = mtile * 16 + (lane & 15);
  const int khi   = (lane >> 4) << 3;                   // A-frag K offset 0/8
  const float* __restrict__ xrow = X + (size_t)mrow * C_DIM;

  v8f acc[16] = {};                                     // 16 o-tiles

#pragma unroll 1
  for (int kc = 0; kc < 8; ++kc) {
    // A 16x32 bf16 fragment (ISA layout): lane<16 -> M=lane, K pairs at
    // {0,2,4,6}+16*{0,1}; lane>=16 -> same M, K pairs offset +8.
    FragBF a;
#pragma unroll
    for (int d = 0; d < 8; ++d) {
      const int k = (kc << 5) + ((d >> 2) << 4) + khi + ((d & 3) << 1);
      const float2 xv = *(const float2*)(xrow + k);
      a.u[d] = pack_bf16(xv.x, xv.y);
    }
#pragma unroll
    for (int ot = 0; ot < 16; ++ot) {
      FragBF b;
      const uint4* bp = (const uint4*)&wlds[ot][kc][lane][0];
      b.q[0] = bp[0];
      b.q[1] = bp[1];
      acc[ot] = __builtin_amdgcn_wmma_f32_16x16x32_bf16(
          false, a.v, false, b.v, (short)0, acc[ot], false, false);
    }
  }

  // Epilogue: BN partial stats + bf16 Y store.
  // D layout: lane<16 -> N=lane, M=vgpr; lane>=16 -> N=lane-16, M=8+vgpr.
#pragma unroll
  for (int ot = 0; ot < 16; ++ot) {
    float ls1 = 0.0f, ls2 = 0.0f;
#pragma unroll
    for (int r = 0; r < 8; ++r) {
      const float y = acc[ot][r];
      ls1 += y;
      ls2 += y * y;
    }
    const int o = (ot << 4) | (lane & 15);
    atomicAdd(&s1[o], ls1);
    atomicAdd(&s2[o], ls2);

    // Pack row pairs (2q, 2q+1) of the tile into one dword:
    // Yws[(mtile*8 + q)*256 + o], q = vgpr-pair + 4*(lane>=16).
    const int qhi = (lane >> 4) << 2;
#pragma unroll
    for (int i = 0; i < 4; ++i) {
      Yws[((size_t)mtile * 8 + (qhi + i)) * C_DIM + o] =
          pack_bf16(acc[ot][2 * i], acc[ot][2 * i + 1]);
    }
  }
  __syncthreads();
  atomicAdd(&stats[tid],          s1[tid]);
  atomicAdd(&stats[C_DIM + tid],  s2[tid]);
}

// ---------------------------------------------------------------------------
// Kernel 2: BN-apply + 4-step LIF + permuted store.
// out[tb, (n%4)*256 + c, n/4] = spike(t=tb/32, b=tb%32, n, c).
// Grid: 2048 blocks x 256 threads; thread = channel c; block = (b, r=n%4,
// 16-wide n/4 block). v state lives in registers across the t loop; spikes
// go through a padded LDS tile so stores are 64B-contiguous runs.
// ---------------------------------------------------------------------------
__global__ __launch_bounds__(256, 1) void snn_bn_lif(
    const unsigned* __restrict__ Yws, const float* __restrict__ stats,
    const float* __restrict__ gamma, const float* __restrict__ beta,
    float* __restrict__ out)
{
  __shared__ float sp[16 * 257];                 // padded: conflict-free
  const int tid = threadIdx.x;                   // channel c
  const int b   = blockIdx.x >> 6;               // 0..31
  const int r   = (blockIdx.x >> 4) & 3;         // n % 4
  const int n4b = (blockIdx.x & 15) << 4;        // base of n/4 block

  const float mean  = stats[tid] * INV_M;
  const float var   = stats[C_DIM + tid] * INV_M - mean * mean;
  const float scale = rsqrtf(var + BN_EPS) * gamma[tid];
  const float shift = beta[tid] - mean * scale;

  float v[16];
#pragma unroll
  for (int j = 0; j < 16; ++j) v[j] = 0.0f;

  for (int t = 0; t < T_STEPS; ++t) {
    const int tb = t * 32 + b;
#pragma unroll
    for (int j = 0; j < 16; ++j) {
      const int n = ((n4b + j) << 2) | r;
      const int m = tb * N_DIM + n;              // GEMM row
      const unsigned dword =
          Yws[((size_t)(m >> 4) * 8 + ((m & 15) >> 1)) * C_DIM + tid];
      const float y = bf16_to_f32((dword >> ((m & 1) << 4)) & 0xFFFFu);
      const float z = y * scale + shift;         // BatchNorm
      const float vj = 0.5f * (v[j] + z);        // LIF: v += (z - v)/tau
      const float s = (vj >= 1.0f) ? 1.0f : 0.0f;
      sp[j * 257 + tid] = s;
      v[j] = vj * (1.0f - s);                    // hard reset
    }
    __syncthreads();
    const size_t obase =
        (size_t)tb * (N_DIM * C_DIM) + (size_t)r * 65536 + n4b;
#pragma unroll
    for (int i = 0; i < 16; ++i) {
      const int cw = (tid >> 4) + (i << 4);      // out channel
      const int pj = tid & 15;                   // n/4 offset
      out[obase + (size_t)cw * 256 + pj] = sp[pj * 257 + cw];
    }
    __syncthreads();
  }
}

extern "C" void kernel_launch(void* const* d_in, const int* in_sizes, int n_in,
                              void* d_out, int out_size, void* d_ws, size_t ws_size,
                              hipStream_t stream) {
  (void)in_sizes; (void)n_in; (void)out_size; (void)ws_size;
  const float* x     = (const float*)d_in[0];
  const float* W     = (const float*)d_in[1];
  const float* gamma = (const float*)d_in[2];
  const float* beta  = (const float*)d_in[3];
  float* out = (float*)d_out;

  float*    stats = (float*)d_ws;                       // 512 floats
  unsigned* Yws   = (unsigned*)((char*)d_ws + 4096);    // 16.8M bf16-pair dwords (67 MB)

  zero_stats<<<2, 256, 0, stream>>>(stats);
  snn_gemm_bn_stats<<<1024, 256, 0, stream>>>(x, W, stats, Yws);
  snn_bn_lif<<<2048, 256, 0, stream>>>(Yws, stats, gamma, beta, out);
}
